// StressXLSTM_20581483282701
// MI455X (gfx1250) — compile-verified, matrix operand order checked by reading, not run on hardware
//
#include <hip/hip_runtime.h>
#include <hip/hip_bf16.h>

// Problem constants (from reference): V=30000, D=128, H=4, HD=32, L=128, B=512, NL=2
#define DD   128
#define HH_  4
#define HD_  32
#define LL   128
#define BB   512
#define NOUT 512   // 4*D gate width

typedef __attribute__((ext_vector_type(16))) __bf16 v16bf;
typedef __attribute__((ext_vector_type(8)))  float  v8f;

union Frag { v16bf v; unsigned u[8]; };

__device__ __forceinline__ unsigned short f2bf(float f) {
  union { float f; unsigned u; } x; x.f = f;
  unsigned r = (x.u + 0x7FFFu + ((x.u >> 16) & 1u)) >> 16;
  return (unsigned short)r;
}

// per-lane K offset for 16-bit A/B fragments (ISA 7.12.2):
// lanes 0-15: vgpr0-3 -> K 0..7, vgpr4-7 -> K 16..23 ; lanes 16-31: K 8..15 / 24..31
__device__ __forceinline__ int kmap(int v, int half) {
  return ((v >> 2) << 4) + (half << 3) + ((v & 3) << 1);
}

// ---- CDNA5 async LDS copy (ASYNCcnt path). Inline asm per bridge doc (portable
// across ROCm7.2 / amdgpu-toolchain). Copies 16B per lane: LDS[%0] = MEM[%1].
__device__ __forceinline__ void async_b128(unsigned lds_byte_addr,
                                           unsigned long long gaddr) {
  asm volatile("global_load_async_to_lds_b128 %0, %1, off"
               :: "v"(lds_byte_addr), "v"(gaddr) : "memory");
}
// copy one 32KB slab (16 batches x 512 f32) with 8 instructions per wave
__device__ __forceinline__ void async_copy_slab(const float* gsrc, float* ldst, int tid) {
  unsigned lbase = (unsigned)(uintptr_t)ldst;       // low 32 bits of shared ptr = LDS offset
  unsigned long long gbase = (unsigned long long)(uintptr_t)gsrc;
  #pragma unroll
  for (int i = 0; i < 8; i++) {
    unsigned off = (unsigned)(tid * 16 + i * 4096);
    async_b128(lbase + off, gbase + off);
  }
}
__device__ __forceinline__ void wait_async8() {
#if __has_builtin(__builtin_amdgcn_s_wait_asynccnt)
  __builtin_amdgcn_s_wait_asynccnt(8);
#else
  asm volatile("s_wait_asynccnt 0x8" ::: "memory");
#endif
}
__device__ __forceinline__ void wait_async0() {
#if __has_builtin(__builtin_amdgcn_s_wait_asynccnt)
  __builtin_amdgcn_s_wait_asynccnt(0);
#else
  asm volatile("s_wait_asynccnt 0x0" ::: "memory");
#endif
}

// ---------------------------------------------------------------- embedding
__global__ void k_embed(const int* __restrict__ ids, const float* __restrict__ emb,
                        float* __restrict__ x) {
  int i = blockIdx.x * 256 + threadIdx.x;          // over B*L*D
  int r = i >> 7, d = i & 127;
  x[i] = emb[(size_t)ids[r] * DD + d];
}

// -------------------------------------------- per-layer weight prep (bf16)
// Wbf: [512][128]  (natural (g,o,d) flatten of Wg[l])
// Rt : [g][h][e][d] bf16 (transpose of Rg[l] (g,h,d,e))
__global__ void k_prep(const float* __restrict__ Wl, const float* __restrict__ Rl,
                       unsigned short* __restrict__ wbf, unsigned short* __restrict__ rt) {
  int i = blockIdx.x * 256 + threadIdx.x;
  if (i < 65536) {
    wbf[i] = f2bf(Wl[i]);
  } else {
    int j  = i - 65536;                 // over 4*4*32*32 = 16384
    int gh = j >> 10;
    int d  = (j >> 5) & 31;
    int e  = j & 31;
    rt[gh * 1024 + e * 32 + d] = f2bf(Rl[j]);
  }
}

// ---------------------------------------------------------------- layernorm
// one wave per row of 128; output bf16
__global__ void k_ln(const float* __restrict__ x, const float* __restrict__ w,
                     const float* __restrict__ b, unsigned short* __restrict__ xn) {
  int row  = blockIdx.x * 8 + (threadIdx.x >> 5);
  int lane = threadIdx.x & 31;
  const float* xr = x + (size_t)row * DD;
  float v0 = xr[lane * 4 + 0], v1 = xr[lane * 4 + 1], v2 = xr[lane * 4 + 2], v3 = xr[lane * 4 + 3];
  float s  = v0 + v1 + v2 + v3;
  float q  = v0 * v0 + v1 * v1 + v2 * v2 + v3 * v3;
  #pragma unroll
  for (int m = 16; m >= 1; m >>= 1) { s += __shfl_xor(s, m, 32); q += __shfl_xor(q, m, 32); }
  float mu  = s * (1.0f / 128.0f);
  float var = q * (1.0f / 128.0f) - mu * mu;
  float rs  = rsqrtf(var + 1e-5f);
  unsigned short* o = xn + (size_t)row * DD + lane * 4;
  #pragma unroll
  for (int i = 0; i < 4; i++) {
    int d = lane * 4 + i;
    float vv = (i == 0 ? v0 : i == 1 ? v1 : i == 2 ? v2 : v3);
    o[i] = f2bf((vv - mu) * rs * w[d] + b[d]);
  }
}

// ------------------------------------------------------ input-projection GEMM
// pre[l_seq][b][j] = sum_d xn[r=(b,l)][d] * Wbf[j][d] + bias[j]
// Block: 128 rows x 64 cols. B slab (64x128 bf16 = 16KB) staged once in LDS;
// per-wave 16x64 tile with all 4 A fragments hoisted before the WMMA chain.
__global__ void __launch_bounds__(256) k_gemm(const unsigned short* __restrict__ xn,
                       const unsigned short* __restrict__ wbf,
                       const float* __restrict__ bias, float* __restrict__ pre) {
  __shared__ uint4 bsl4[1024];                       // 64 rows x 128 bf16
  const int tid = threadIdx.x;
  const int w = tid >> 5, lane = tid & 31;
  const int half = lane >> 4, idx = lane & 15;
  const int m0 = blockIdx.x * 128 + w * 16;
  const int n0 = blockIdx.y * 64;

  // cooperative stage of B slab
  const uint4* bsrc = (const uint4*)(wbf + (size_t)n0 * DD);
  #pragma unroll
  for (int k = 0; k < 4; k++) bsl4[tid + k * 256] = bsrc[tid + k * 256];

  // hoist all A fragments (full K=128) — global loads issue before any WMMA
  Frag a[4];
  #pragma unroll
  for (int kc = 0; kc < 4; kc++)
    #pragma unroll
    for (int v = 0; v < 8; v++)
      a[kc].u[v] = *(const unsigned*)(xn + (size_t)(m0 + idx) * DD + kc * 32 + kmap(v, half));

  __syncthreads();
  const unsigned short* bsl = (const unsigned short*)bsl4;

  v8f C[4] = {};
  #pragma unroll
  for (int kc = 0; kc < 4; kc++) {
    #pragma unroll
    for (int t = 0; t < 4; t++) {
      Frag bmat;
      #pragma unroll
      for (int v = 0; v < 8; v++)
        bmat.u[v] = *(const unsigned*)&bsl[(t * 16 + idx) * DD + kc * 32 + kmap(v, half)];
      C[t] = __builtin_amdgcn_wmma_f32_16x16x32_bf16(false, a[kc].v, false, bmat.v,
                                                     (short)0, C[t], false, false);
    }
  }
  #pragma unroll
  for (int t = 0; t < 4; t++) {
    #pragma unroll
    for (int v = 0; v < 8; v++) {
      int r  = m0 + half * 8 + v;          // r = b*L + l
      int j  = n0 + t * 16 + idx;
      int ls = r & (LL - 1);
      int bb = r >> 7;
      pre[((size_t)ls * BB + bb) * NOUT + j] = C[t][v] + bias[j];
    }
  }
}

// ---------------------------------------------------------- recurrent scan
// 32 blocks x 16-batch tiles; 8 waves: wave w -> gate g=w&3, head-pair hh=w>>2.
// R fragments live in registers for all 128 steps. The per-step 32KB `pre`
// slab is double-buffered in LDS via async DMA (ASYNCcnt) one step ahead.
// Dynamic LDS: 2*32KB preBuf + 32KB recs + 4KB hbf = 100KB (WGP has 320KB).
__global__ void __launch_bounds__(256) k_scan(const float* __restrict__ pre,
                       const unsigned short* __restrict__ rt,
                       float* __restrict__ hs) {
  extern __shared__ char smem[];
  float*          preBuf = (float*)smem;                         // [2][16*512]
  float*          recs   = (float*)(smem + 65536);               // [16*512]
  unsigned short* hbf    = (unsigned short*)(smem + 65536 + 32768); // [16*128]

  const int tid  = threadIdx.x;
  const int w    = tid >> 5, lane = tid & 31;
  const int half = lane >> 4, idx = lane & 15;
  const int g    = w & 3, hh = w >> 2;
  const int b0   = blockIdx.x * 16;

  // preload R fragments (fixed across steps): [head-local][ntile]
  Frag bf[2][2];
  #pragma unroll
  for (int hl = 0; hl < 2; hl++) {
    int head = hh * 2 + hl;
    #pragma unroll
    for (int nt = 0; nt < 2; nt++)
      #pragma unroll
      for (int v = 0; v < 8; v++)
        bf[hl][nt].u[v] = *(const unsigned*)(rt + ((size_t)(g * HH_ + head) * 32
                                                  + nt * 16 + idx) * 32 + kmap(v, half));
  }

  // kick off DMA of step-0 slab into buffer 0
  async_copy_slab(pre + (size_t)b0 * NOUT, preBuf, tid);

  // zero h state + per-thread cell state
  #pragma unroll
  for (int q = 0; q < 8; q++) hbf[q * 256 + tid] = 0;
  float cs[8], ns[8], ms[8];
  #pragma unroll
  for (int q = 0; q < 8; q++) { cs[q] = 0.f; ns[q] = 0.f; ms[q] = 0.f; }
  __syncthreads();

  for (int l = 0; l < LL; l++) {
    const float* bufCur = preBuf + (l & 1) * 8192;
    // prefetch next step's slab while we do WMMA + gating on this one
    if (l + 1 < LL)
      async_copy_slab(pre + (size_t)(l + 1) * BB * NOUT + (size_t)b0 * NOUT,
                      preBuf + ((l + 1) & 1) * 8192, tid);

    // rec = h x R (bf16 WMMA, f32 acc)
    Frag af[2];
    #pragma unroll
    for (int hl = 0; hl < 2; hl++) {
      int head = hh * 2 + hl;
      #pragma unroll
      for (int v = 0; v < 8; v++)
        af[hl].u[v] = *(const unsigned*)&hbf[idx * DD + head * 32 + kmap(v, half)];
    }
    v8f Cr[2][2] = {};
    #pragma unroll
    for (int hl = 0; hl < 2; hl++)
      #pragma unroll
      for (int nt = 0; nt < 2; nt++)
        Cr[hl][nt] = __builtin_amdgcn_wmma_f32_16x16x32_bf16(false, af[hl].v, false,
                          bf[hl][nt].v, (short)0, Cr[hl][nt], false, false);
    #pragma unroll
    for (int hl = 0; hl < 2; hl++) {
      int head = hh * 2 + hl;
      #pragma unroll
      for (int nt = 0; nt < 2; nt++)
        #pragma unroll
        for (int v = 0; v < 8; v++)
          recs[(half * 8 + v) * NOUT + g * DD + head * 32 + nt * 16 + idx] = Cr[hl][nt][v];
    }
    // current slab (issued one step ago) must have landed: <=8 outstanding
    // means the oldest 8 (this step's) are done; last step waits to 0.
    if (l + 1 < LL) wait_async8(); else wait_async0();
    __syncthreads();

    // gating: each thread owns 8 fixed (b,d) slots -> c,n,m stay in registers
    #pragma unroll
    for (int q = 0; q < 8; q++) {
      int s  = q * 256 + tid;
      int bb = s >> 7, d = s & 127;
      const float* pr = bufCur + bb * NOUT;
      float it = pr[0 * DD + d] + recs[bb * NOUT + 0 * DD + d];
      float ft = pr[1 * DD + d] + recs[bb * NOUT + 1 * DD + d];
      float zt = pr[2 * DD + d] + recs[bb * NOUT + 2 * DD + d];
      float ot = pr[3 * DD + d] + recs[bb * NOUT + 3 * DD + d];
      float mn = fmaxf(ft + ms[q], it);
      float iv = expf(it - mn);
      float fv = expf(ft + ms[q] - mn);
      cs[q] = fv * cs[q] + iv * tanhf(zt);
      ns[q] = fv * ns[q] + iv;
      ms[q] = mn;
      float hv = (1.0f / (1.0f + expf(-ot))) * cs[q] / ns[q];
      hs[((size_t)l * BB + b0 + bb) * DD + d] = hv;
      hbf[bb * DD + d] = f2bf(hv);
    }
    __syncthreads();
  }
}

// ------------------------------------------- head-wise groupnorm + residual
__global__ void k_gnorm(const float* __restrict__ hs, const float* __restrict__ gw,
                        float* __restrict__ x) {
  int row  = blockIdx.x * 8 + (threadIdx.x >> 5);   // row = b*L + l
  int lane = threadIdx.x & 31;
  int b = row >> 7, l = row & (LL - 1);
  const float* hr = hs + ((size_t)l * BB + b) * DD;
  float v[4], s = 0.f, q = 0.f;
  #pragma unroll
  for (int i = 0; i < 4; i++) { v[i] = hr[lane * 4 + i]; s += v[i]; q += v[i] * v[i]; }
  #pragma unroll
  for (int m = 4; m >= 1; m >>= 1) { s += __shfl_xor(s, m, 8); q += __shfl_xor(q, m, 8); }
  float mu  = s * (1.0f / 32.0f);
  float var = q * (1.0f / 32.0f) - mu * mu;
  float rs  = rsqrtf(var + 1e-5f);
  float* xr = x + (size_t)row * DD;
  #pragma unroll
  for (int i = 0; i < 4; i++) {
    int d = lane * 4 + i;
    xr[d] += (v[i] - mu) * rs * gw[d];
  }
}

// --------------------------------------------------- mean pool + tiny MLP head
__global__ void k_head(const float* __restrict__ x, const float* __restrict__ w1,
                       const float* __restrict__ b1, const float* __restrict__ w2,
                       const float* __restrict__ b2, float* __restrict__ out) {
  __shared__ float pool[DD];
  __shared__ float hid[64];
  int b = blockIdx.x, t = threadIdx.x;
  float s = 0.f;
  for (int l = 0; l < LL; l++) s += x[((size_t)b * LL + l) * DD + t];
  pool[t] = s * (1.0f / (float)LL);
  __syncthreads();
  if (t < 64) {
    float a = b1[t];
    #pragma unroll 4
    for (int d = 0; d < DD; d++) a += pool[d] * w1[t * DD + d];
    hid[t] = fmaxf(a, 0.f);
  }
  __syncthreads();
  if (t < 2) {
    float a = b2[t];
    #pragma unroll
    for (int j = 0; j < 64; j++) a += hid[j] * w2[t * 64 + j];
    out[b * 2 + t] = a;
  }
}

extern "C" void kernel_launch(void* const* d_in, const int* in_sizes, int n_in,
                              void* d_out, int out_size, void* d_ws, size_t ws_size,
                              hipStream_t stream) {
  const int*   ids  = (const int*)  d_in[0];
  const float* emb  = (const float*)d_in[1];
  const float* ln_w = (const float*)d_in[2];
  const float* ln_b = (const float*)d_in[3];
  const float* Wg   = (const float*)d_in[4];   // (2,4,128,128)
  const float* Rg   = (const float*)d_in[5];   // (2,4,4,32,32)
  const float* bg   = (const float*)d_in[6];   // (2,512)
  const float* gn_w = (const float*)d_in[7];
  const float* w1   = (const float*)d_in[8];
  const float* b1   = (const float*)d_in[9];
  const float* w2   = (const float*)d_in[10];
  const float* b2   = (const float*)d_in[11];

  char* ws = (char*)d_ws;
  float*          x   = (float*)ws;                                  //  33,554,432 B
  unsigned short* xn  = (unsigned short*)(ws + 33554432ull);         // +16,777,216
  float*          pre = (float*)(ws + 50331648ull);                  // +134,217,728
  float*          hs  = (float*)(ws + 184549376ull);                 // +33,554,432
  unsigned short* wbf = (unsigned short*)(ws + 218103808ull);        // +131,072
  unsigned short* rt  = (unsigned short*)(ws + 218234880ull);        // +32,768

  const int ROWS = BB * LL;                    // 65536
  const size_t scanLds = 65536 + 32768 + 4096; // 100KB dynamic LDS

  k_embed<<<ROWS * DD / 256, 256, 0, stream>>>(ids, emb, x);

  for (int l = 0; l < 2; l++) {
    k_prep<<<(65536 + 16384) / 256, 256, 0, stream>>>(Wg + (size_t)l * 65536,
                                                      Rg + (size_t)l * 16384, wbf, rt);
    k_ln<<<ROWS / 8, 256, 0, stream>>>(x, ln_w + l * DD, ln_b + l * DD, xn);
    k_gemm<<<dim3(ROWS / 128, NOUT / 64), 256, 0, stream>>>(xn, wbf, bg + l * NOUT, pre);
    k_scan<<<BB / 16, 256, scanLds, stream>>>(pre, rt, hs);
    k_gnorm<<<ROWS / 8, 256, 0, stream>>>(hs, gn_w + l * DD, x);
  }

  k_head<<<BB, DD, 0, stream>>>(x, w1, b1, w2, b2, (float*)d_out);
}
